// se3ACN_3917010173962
// MI455X (gfx1250) — matrix operand order repaired and checked by reading, not compiled, not run on hardware
//
#include <hip/hip_runtime.h>
#include <cstdint>
#include <cstddef>

// ---------------------------------------------------------------------------
// Problem constants
// ---------------------------------------------------------------------------
#define BB 16
#define NN 64
#define PAIRS (BB * NN * NN)      // 65536
#define EMB 32
#define CLOUD 24
#define RH 100
#define RHPAD 128                 // K padded (col 100 = 1.0 bias lane)
#define NPAD 112                  // 7 tiles of 16 for the 100-wide hidden layers
#define NHID 4
#define LU 72                     // 3 * 24
#define LUPAD 80                  // 5 tiles of 16
#define CFEAT 219
#define CFPAD 224

#define WPACK_ELEMS (7 * 4 * 512)       // per-layer packed weights, bf16
#define WEFF_ELEMS  (5 * 4 * 512)       // per-(b,j) packed Weff, bf16

typedef __bf16 bf16_t;
typedef __attribute__((ext_vector_type(16))) __bf16 v16bf;
typedef __attribute__((ext_vector_type(8)))  float  v8f;

// ---------------------------------------------------------------------------
// Math helpers
// ---------------------------------------------------------------------------
__device__ __forceinline__ float sp5f(float x) {        // softplus(5x)/5
    float t = 5.0f * x;
    return (t > 30.0f) ? x : log1pf(__expf(t)) * 0.2f;
}
__device__ __forceinline__ float softplusf(float x) {
    return (x > 30.0f) ? x : log1pf(__expf(x));
}
__device__ __forceinline__ float sigmoidf(float x) {
    return 1.0f / (1.0f + __expf(-x));
}

// ---------------------------------------------------------------------------
// CDNA5 async global->LDS copy (ASYNCcnt path) + wait
// ---------------------------------------------------------------------------
__device__ __forceinline__ void async_copy_b128(uint32_t lds_byte_off, const void* gaddr) {
    asm volatile("global_load_async_to_lds_b128 %0, %1, off"
                 :: "v"(lds_byte_off), "v"(gaddr)
                 : "memory");
}
__device__ __forceinline__ void wait_async0() {
    asm volatile("s_wait_asynccnt 0x0" ::: "memory");
}

// A fragment (16x32 bf16, M x K), row-major source, contiguous K order:
// lanes 0-15 hold M=lane, K 0..15; lanes 16-31 hold K 16..31.
__device__ __forceinline__ v16bf load_a_frag(const bf16_t* row_base, int lane) {
    int half = lane >> 4;
    const bf16_t* p0 = row_base + half * 16;  // 16 contiguous bf16 (32B)
    v16bf a;
#pragma unroll
    for (int e = 0; e < 8; ++e) a[e] = p0[e];
#pragma unroll
    for (int e = 0; e < 8; ++e) a[8 + e] = p0[8 + e];
    return a;
}
// B fragment from a pre-packed buffer (global or LDS): 16 contiguous bf16 / lane.
__device__ __forceinline__ v16bf load_b_packed(const bf16_t* p, int lane) {
    const bf16_t* q = p + lane * 16;
    v16bf b;
#pragma unroll
    for (int e = 0; e < 16; ++e) b[e] = q[e];
    return b;
}

// ---------------------------------------------------------------------------
// K0: zero a bf16 region (WeffPack padding)
// ---------------------------------------------------------------------------
__global__ void k_zero_bf16(bf16_t* p, size_t n) {
    size_t i = (size_t)blockIdx.x * blockDim.x + threadIdx.x;
    size_t stride = (size_t)gridDim.x * blockDim.x;
    for (; i < n; i += stride) p[i] = (bf16_t)0.0f;
}

// ---------------------------------------------------------------------------
// K1: pack the 4 hidden-layer weights (100x100 f32 + bias) into WMMA-B layout,
// padded to 128(K) x 112(N); K-row 100 carries the bias.
// packed index: layer*WPACK_ELEMS + (nt*4+kt)*512 + lane*16 + e ;
// K = kt*32 + (lane>>4)*16 + e ; N = nt*16 + (lane&15)
// ---------------------------------------------------------------------------
__global__ void k_pack_wh(const float* Wh, const float* bh, bf16_t* wpack) {
    int idx = blockIdx.x * blockDim.x + threadIdx.x;
    const int total = NHID * WPACK_ELEMS;
    if (idx >= total) return;
    int e = idx & 15;
    int lane = (idx >> 4) & 31;
    int kt = (idx >> 9) & 3;
    int nt = (idx >> 11) % 7;
    int layer = idx / WPACK_ELEMS;
    int K = kt * 32 + (lane >> 4) * 16 + e;
    int Ncol = nt * 16 + (lane & 15);
    float v = 0.0f;
    if (Ncol < RH) {
        if (K < RH)        v = Wh[(size_t)layer * RH * RH + (size_t)K * RH + Ncol];
        else if (K == RH)  v = bh[layer * RH + Ncol];
    }
    wpack[idx] = (bf16_t)v;
}

// ---------------------------------------------------------------------------
// K2: per-pair geometry + cosine basis + first radial layer (K=3, VALU).
// Writes h0 [P x 128] bf16 (col100=1, pads 0), Y [P x 9] f32, mask [P] f32.
// ---------------------------------------------------------------------------
__global__ void k_stageA(const float* __restrict__ xyz,
                         const float* __restrict__ W0, const float* __restrict__ b0,
                         bf16_t* __restrict__ h0, float* __restrict__ Ybuf,
                         float* __restrict__ maskf) {
    int p = blockIdx.x;
    int b = p >> 12, i = (p >> 6) & 63, j = p & 63;
    int t = threadIdx.x;
    const float* xi = xyz + (size_t)(b * NN + i) * 3;
    const float* xj = xyz + (size_t)(b * NN + j) * 3;
    float rx = xj[0] - xi[0], ry = xj[1] - xi[1], rz = xj[2] - xi[2];
    float r = sqrtf(rx * rx + ry * ry + rz * rz + 1e-12f);
    float inv = 1.0f / r;
    float bas[3];
#pragma unroll
    for (int k = 0; k < 3; ++k) {
        float d = r - (float)k;
        float c = __cosf(1.5707963267948966f * d);
        bas[k] = (fabsf(d) < 1.0f) ? c * c : 0.0f;
    }
    bf16_t* hrow = h0 + (size_t)p * RHPAD;
    if (t < RH) {
        float acc = b0[t];
#pragma unroll
        for (int k = 0; k < 3; ++k) acc += bas[k] * W0[k * RH + t];
        hrow[t] = (bf16_t)sp5f(acc);
    } else if (t == RH) {
        hrow[t] = (bf16_t)1.0f;                 // bias lane
    } else if (t < RHPAD) {
        hrow[t] = (bf16_t)0.0f;
    }
    if (t == 0) {
        float ux = rx * inv, uy = ry * inv, uz = rz * inv;
        float* Y = Ybuf + (size_t)p * 9;
        const float c0 = 0.28209479177387814f, c1 = 0.4886025119029199f;
        const float c2 = 1.0925484305920792f, c20 = 0.31539156525252005f, c22 = 0.5462742152960396f;
        Y[0] = c0;
        Y[1] = c1 * uy; Y[2] = c1 * uz; Y[3] = c1 * ux;
        Y[4] = c2 * ux * uy; Y[5] = c2 * uy * uz;
        Y[6] = c20 * (3.0f * uz * uz - 1.0f);
        Y[7] = c2 * ux * uz; Y[8] = c22 * (ux * ux - uy * uy);
        maskf[p] = (r < 2.0f && i != j) ? 1.0f : 0.0f;
    }
}

// K2b: neighbor counts per (b,i)
__global__ void k_nnb(const float* __restrict__ maskf, float* __restrict__ nnb) {
    int bi = blockIdx.x * blockDim.x + threadIdx.x;
    if (bi >= BB * NN) return;
    float s = 0.0f;
    for (int j = 0; j < NN; ++j) s += maskf[(size_t)bi * NN + j];
    nnb[bi] = fmaxf(s, 1.0f);
}

// ---------------------------------------------------------------------------
// K3: hidden radial layer — WMMA GEMM [65536 x 100] @ [100 x 100] (+bias,sp5).
// Packed weights (28 KB) staged to LDS via global_load_async_to_lds_b128 and
// shared by all 8 waves; B fragments fed from ds_load.
// ---------------------------------------------------------------------------
__global__ void __launch_bounds__(256) k_hidden(const bf16_t* __restrict__ hin,
                                                const bf16_t* __restrict__ wpack,
                                                bf16_t* __restrict__ hout) {
    __shared__ __align__(16) bf16_t sW[WPACK_ELEMS];    // 28672 B
    {
        uint32_t base = (uint32_t)(uintptr_t)(&sW[0]);
        const char* g = (const char*)wpack;
#pragma unroll
        for (int c = 0; c < 7; ++c) {                   // 1792 x 16B chunks / 256 thr
            int idx = threadIdx.x + c * 256;
            async_copy_b128(base + idx * 16, g + (size_t)idx * 16);
        }
        wait_async0();
        __syncthreads();
    }

    int lane = threadIdx.x & 31, wv = threadIdx.x >> 5;
    int half = lane >> 4, mrow = lane & 15;
    int rowbase = blockIdx.x * 128 + wv * 16;
    v8f zero = {0.f, 0.f, 0.f, 0.f, 0.f, 0.f, 0.f, 0.f};
    v8f acc[7];
#pragma unroll
    for (int nt = 0; nt < 7; ++nt) acc[nt] = zero;

#pragma unroll
    for (int kt = 0; kt < 4; ++kt) {
        const bf16_t* pa = hin + (size_t)(rowbase + mrow) * RHPAD + kt * 32;
        v16bf a = load_a_frag(pa, lane);
        v16bf bf[7];
#pragma unroll
        for (int nt = 0; nt < 7; ++nt)
            bf[nt] = load_b_packed(&sW[(nt * 4 + kt) * 512], lane);
#pragma unroll
        for (int nt = 0; nt < 7; ++nt)
            acc[nt] = __builtin_amdgcn_wmma_f32_16x16x32_bf16(
                false, a, false, bf[nt], (short)0, acc[nt], false, false);
    }

#pragma unroll
    for (int nt = 0; nt < 7; ++nt) {
        int n = nt * 16 + mrow;
#pragma unroll
        for (int r = 0; r < 8; ++r) {
            int row = rowbase + half * 8 + r;
            bf16_t* o = hout + (size_t)row * RHPAD;
            if (n < RH)       o[n] = (bf16_t)sp5f(acc[nt][r]);
            else if (n == RH) o[n] = (bf16_t)1.0f;
            else              o[n] = (bf16_t)0.0f;
        }
    }
#pragma unroll
    for (int r = 0; r < 8; ++r) {  // pad cols 112..127
        int row = rowbase + half * 8 + r;
        hout[(size_t)row * RHPAD + NPAD + mrow] = (bf16_t)0.0f;
    }
}

// ---------------------------------------------------------------------------
// K4: Weff[b,j,h,lu] = sum_v Wout[h,lu,v]*emb[Z[b,j],v]  (h==100 row <- bout)
// stored directly in packed WMMA-B layout [bj][5 ntiles][4 ktiles][32 lanes][16]
// ---------------------------------------------------------------------------
__global__ void k_stageC(const int* __restrict__ Z, const float* __restrict__ emb,
                         const float* __restrict__ Wout, const float* __restrict__ bout,
                         bf16_t* __restrict__ weffpack) {
    int idx = blockIdx.x * blockDim.x + threadIdx.x;
    const int total = BB * NN * (RH + 1) * LU;
    if (idx >= total) return;
    int lu = idx % LU;
    int tmp = idx / LU;
    int h = tmp % (RH + 1);
    int bj = tmp / (RH + 1);
    const float* f = emb + (size_t)Z[bj] * EMB;
    const float* wsrc = (h < RH) ? (Wout + (size_t)h * (3 * CLOUD * EMB) + (size_t)lu * EMB)
                                 : (bout + (size_t)lu * EMB);
    float acc = 0.0f;
#pragma unroll
    for (int v = 0; v < EMB; ++v) acc += wsrc[v] * f[v];
    int kt = h >> 5, halfk = (h >> 4) & 1, e = h & 15;
    int nt = lu >> 4, lanep = halfk * 16 + (lu & 15);
    weffpack[(size_t)bj * WEFF_ELEMS + (size_t)(nt * 4 + kt) * 512 + lanep * 16 + e] = (bf16_t)acc;
}

// ---------------------------------------------------------------------------
// K5: batched T GEMM — per (b,j): T[64 x 72] = h4[b,:,j,:] @ Weff[b,j] ; masked.
// 20 KB Weff block async-staged to LDS, shared by the 4 waves.
// ---------------------------------------------------------------------------
__global__ void __launch_bounds__(128) k_stageD(const bf16_t* __restrict__ h4,
                                                const bf16_t* __restrict__ weffpack,
                                                const float* __restrict__ maskf,
                                                float* __restrict__ Tbuf) {
    __shared__ __align__(16) bf16_t sW[WEFF_ELEMS];     // 20480 B
    int bj = blockIdx.x;
    int b = bj >> 6, j = bj & 63;
    {
        uint32_t base = (uint32_t)(uintptr_t)(&sW[0]);
        const char* g = (const char*)(weffpack + (size_t)bj * WEFF_ELEMS);
#pragma unroll
        for (int c = 0; c < 10; ++c) {                  // 1280 x 16B chunks / 128 thr
            int idx = threadIdx.x + c * 128;
            async_copy_b128(base + idx * 16, g + (size_t)idx * 16);
        }
        wait_async0();
        __syncthreads();
    }

    int lane = threadIdx.x & 31, wv = threadIdx.x >> 5;
    int half = lane >> 4, mrow = lane & 15;
    v8f zero = {0.f, 0.f, 0.f, 0.f, 0.f, 0.f, 0.f, 0.f};
    v8f acc[5];
#pragma unroll
    for (int nt = 0; nt < 5; ++nt) acc[nt] = zero;

#pragma unroll
    for (int kt = 0; kt < 4; ++kt) {
        int irow = wv * 16 + mrow;
        const bf16_t* pa = h4 + ((size_t)((b * NN + irow) * NN + j)) * RHPAD + kt * 32;
        v16bf a = load_a_frag(pa, lane);
        v16bf bf[5];
#pragma unroll
        for (int nt = 0; nt < 5; ++nt)
            bf[nt] = load_b_packed(&sW[(nt * 4 + kt) * 512], lane);
#pragma unroll
        for (int nt = 0; nt < 5; ++nt)
            acc[nt] = __builtin_amdgcn_wmma_f32_16x16x32_bf16(
                false, a, false, bf[nt], (short)0, acc[nt], false, false);
    }

#pragma unroll
    for (int nt = 0; nt < 5; ++nt) {
        int lu = nt * 16 + mrow;
        if (lu >= LU) continue;
#pragma unroll
        for (int r = 0; r < 8; ++r) {
            int irow = wv * 16 + half * 8 + r;
            size_t p = (size_t)((b * NN + irow) * NN + j);
            Tbuf[p * LU + lu] = acc[nt][r] * maskf[p];
        }
    }
}

// ---------------------------------------------------------------------------
// K6: SH contraction per (b,i): f1[u,m] = sum_j T[bij,l*24+u]*Y[bij,m]/sqrt(2l+1)
// then /sqrt(nnb); assembles hfeat = [body23(3) | f1(216)] padded to 224.
// ---------------------------------------------------------------------------
__global__ void k_stageE(const float* __restrict__ Tbuf, const float* __restrict__ Ybuf,
                         const float* __restrict__ nnb, const float* __restrict__ body23,
                         float* __restrict__ hfeat) {
    int bi = blockIdx.x;
    int t = threadIdx.x;
    float* hrow = hfeat + (size_t)bi * CFPAD;
    if (t < 3) { hrow[t] = body23[(size_t)bi * 3 + t]; return; }
    if (t >= CFPAD) return;
    if (t >= CFEAT) { hrow[t] = 0.0f; return; }
    int c = t - 3;  // 0..215
    int l, u, m;
    if (c < 24)      { l = 0; u = c;            m = 0; }
    else if (c < 96) { l = 1; u = (c - 24) / 3; m = 1 + (c - 24) % 3; }
    else             { l = 2; u = (c - 96) / 5; m = 4 + (c - 96) % 5; }
    float norm = rsqrtf(2.0f * (float)l + 1.0f);
    float acc = 0.0f;
    size_t base = (size_t)bi * NN;
    for (int j = 0; j < NN; ++j) {
        size_t p = base + j;
        acc += Tbuf[p * LU + l * CLOUD + u] * Ybuf[p * 9 + m];
    }
    hrow[t] = acc * norm * rsqrtf(nnb[bi]);
}

// ---------------------------------------------------------------------------
// Tail: residual linear -> BN stats -> ReLU+concat+sum -> col MLP -> BN -> out
// ---------------------------------------------------------------------------
__global__ void k_res_linear(const float* __restrict__ hfeat, const float* __restrict__ W,
                             const float* __restrict__ bvec, float* __restrict__ lin) {
    int idx = blockIdx.x * blockDim.x + threadIdx.x;
    if (idx >= BB * NN * CFEAT) return;
    int col = idx % CFEAT, row = idx / CFEAT;
    const float* hr = hfeat + (size_t)row * CFPAD;
    float acc = bvec[col];
    for (int k = 0; k < CFEAT; ++k) acc += hr[k] * W[(size_t)k * CFEAT + col];
    lin[(size_t)row * CFPAD + col] = acc;
}

__global__ void k_bn_stats(const float* __restrict__ lin, float* __restrict__ mv) {
    int c = threadIdx.x;
    if (c >= CFEAT) return;
    float s = 0.0f, s2 = 0.0f;
    for (int row = 0; row < BB * NN; ++row) {
        float v = lin[(size_t)row * CFPAD + c];
        s += v; s2 += v * v;
    }
    float mean = s * (1.0f / (BB * NN));
    mv[c] = mean;
    mv[256 + c] = s2 * (1.0f / (BB * NN)) - mean * mean;
}

__global__ void k_collate(const float* __restrict__ hfeat, const float* __restrict__ lin,
                          const float* __restrict__ mv, const float* __restrict__ gamma,
                          const float* __restrict__ beta, float* __restrict__ xsum) {
    int idx = blockIdx.x * blockDim.x + threadIdx.x;
    if (idx >= BB * 2 * CFEAT) return;
    int c = idx % (2 * CFEAT), b = idx / (2 * CFEAT);
    float s = 0.0f;
    for (int i = 0; i < NN; ++i) {
        int row = b * NN + i;
        if (c < CFEAT) {
            s += hfeat[(size_t)row * CFPAD + c];
        } else {
            int cc = c - CFEAT;
            float v = gamma[cc] * (lin[(size_t)row * CFPAD + cc] - mv[cc]) *
                          rsqrtf(mv[256 + cc] + 1e-5f) + beta[cc];
            s += fmaxf(v, 0.0f);
        }
    }
    xsum[idx] = s;
}

__global__ void k_col_linear(const float* __restrict__ xsum, const float* __restrict__ W,
                             const float* __restrict__ bvec, float* __restrict__ xc) {
    int idx = blockIdx.x * blockDim.x + threadIdx.x;
    if (idx >= BB * 128) return;
    int o = idx % 128, b = idx / 128;
    float acc = bvec[o];
    for (int k = 0; k < 2 * CFEAT; ++k) acc += xsum[(size_t)b * (2 * CFEAT) + k] * W[(size_t)k * 128 + o];
    xc[idx] = softplusf(acc);
}

__global__ void k_bn2_stats(const float* __restrict__ xc, float* __restrict__ mv2) {
    int c = threadIdx.x;
    if (c >= 128) return;
    float s = 0.0f, s2 = 0.0f;
    for (int b = 0; b < BB; ++b) {
        float v = xc[(size_t)b * 128 + c];
        s += v; s2 += v * v;
    }
    float mean = s * (1.0f / BB);
    mv2[c] = mean;
    mv2[128 + c] = s2 * (1.0f / BB) - mean * mean;
}

__global__ void k_final(const float* __restrict__ xc, const float* __restrict__ mv2,
                        const float* __restrict__ g, const float* __restrict__ bt,
                        const float* __restrict__ outW, const float* __restrict__ outB,
                        float* __restrict__ out) {
    int b = threadIdx.x;
    if (b >= BB) return;
    float acc = outB[0];
    for (int k = 0; k < 128; ++k) {
        float v = g[k] * (xc[(size_t)b * 128 + k] - mv2[k]) * rsqrtf(mv2[128 + k] + 1e-5f) + bt[k];
        acc += softplusf(v) * outW[k];
    }
    out[b] = sigmoidf(acc);
}

// ---------------------------------------------------------------------------
// Host-side launch sequencing (graph-capture safe: kernels only, all on stream)
// ---------------------------------------------------------------------------
static inline char* align_up(char* p) {
    return (char*)(((uintptr_t)p + 255u) & ~(uintptr_t)255u);
}

extern "C" void kernel_launch(void* const* d_in, const int* in_sizes, int n_in,
                              void* d_out, int out_size, void* d_ws, size_t ws_size,
                              hipStream_t stream) {
    const float* xyz    = (const float*)d_in[0];
    const int*   Z      = (const int*)  d_in[1];
    const float* body23 = (const float*)d_in[2];
    const float* emb    = (const float*)d_in[3];
    const float* W0     = (const float*)d_in[4];
    const float* b0     = (const float*)d_in[5];
    const float* Wh     = (const float*)d_in[6];
    const float* bh     = (const float*)d_in[7];
    const float* Wout   = (const float*)d_in[8];
    const float* bout   = (const float*)d_in[9];
    const float* res_W  = (const float*)d_in[10];
    const float* res_b  = (const float*)d_in[11];
    const float* res_g  = (const float*)d_in[12];
    const float* res_bt = (const float*)d_in[13];
    const float* col_W  = (const float*)d_in[14];
    const float* col_b  = (const float*)d_in[15];
    const float* bn_g   = (const float*)d_in[16];
    const float* bn_bt  = (const float*)d_in[17];
    const float* out_W  = (const float*)d_in[18];
    const float* out_b  = (const float*)d_in[19];
    float* out = (float*)d_out;

    char* p = (char*)d_ws;
    bf16_t* hA = (bf16_t*)p;            p = align_up(p + (size_t)PAIRS * RHPAD * sizeof(bf16_t));
    bf16_t* hB = (bf16_t*)p;            p = align_up(p + (size_t)PAIRS * RHPAD * sizeof(bf16_t));
    bf16_t* wpack = (bf16_t*)p;         p = align_up(p + (size_t)NHID * WPACK_ELEMS * sizeof(bf16_t));
    bf16_t* weffpack = (bf16_t*)p;      p = align_up(p + (size_t)BB * NN * WEFF_ELEMS * sizeof(bf16_t));
    float* Ybuf = (float*)p;            p = align_up(p + (size_t)PAIRS * 9 * sizeof(float));
    float* maskf = (float*)p;           p = align_up(p + (size_t)PAIRS * sizeof(float));
    float* nnb = (float*)p;             p = align_up(p + (size_t)BB * NN * sizeof(float));
    float* Tbuf = (float*)p;            p = align_up(p + (size_t)PAIRS * LU * sizeof(float));
    float* hfeat = (float*)p;           p = align_up(p + (size_t)BB * NN * CFPAD * sizeof(float));
    float* lin = (float*)p;             p = align_up(p + (size_t)BB * NN * CFPAD * sizeof(float));
    float* mv = (float*)p;              p = align_up(p + 512 * sizeof(float));
    float* xsum = (float*)p;            p = align_up(p + (size_t)BB * 2 * CFEAT * sizeof(float));
    float* xc = (float*)p;              p = align_up(p + (size_t)BB * 128 * sizeof(float));
    float* mv2 = (float*)p;             p = align_up(p + 256 * sizeof(float));
    (void)ws_size; (void)n_in; (void)in_sizes; (void)out_size;

    // 0. zero Weff packed buffer (padding rows/cols must be 0)
    {
        size_t n = (size_t)BB * NN * WEFF_ELEMS;
        k_zero_bf16<<<1024, 256, 0, stream>>>(weffpack, n);
    }
    // 1. pack hidden weights
    {
        int total = NHID * WPACK_ELEMS;
        k_pack_wh<<<(total + 255) / 256, 256, 0, stream>>>(Wh, bh, wpack);
    }
    // 2. geometry + basis + first radial layer
    k_stageA<<<PAIRS, 128, 0, stream>>>(xyz, W0, b0, hA, Ybuf, maskf);
    k_nnb<<<(BB * NN + 255) / 256, 256, 0, stream>>>(maskf, nnb);
    // 3. four hidden layers (WMMA, LDS-staged weights), ping-pong; result in hA
    k_hidden<<<PAIRS / 128, 256, 0, stream>>>(hA, wpack + 0 * WPACK_ELEMS, hB);
    k_hidden<<<PAIRS / 128, 256, 0, stream>>>(hB, wpack + 1 * WPACK_ELEMS, hA);
    k_hidden<<<PAIRS / 128, 256, 0, stream>>>(hA, wpack + 2 * WPACK_ELEMS, hB);
    k_hidden<<<PAIRS / 128, 256, 0, stream>>>(hB, wpack + 3 * WPACK_ELEMS, hA);
    // 4. Weff = Wout contracted with per-atom features (bias row h==100 <- bout)
    {
        int total = BB * NN * (RH + 1) * LU;
        k_stageC<<<(total + 255) / 256, 256, 0, stream>>>(Z, emb, Wout, bout, weffpack);
    }
    // 5. T = h4 @ Weff per (b, j) (WMMA, LDS-staged Weff), masked
    k_stageD<<<BB * NN, 128, 0, stream>>>(hA, weffpack, maskf, Tbuf);
    // 6. spherical-harmonic contraction + neighbor norm + feature assembly
    k_stageE<<<BB * NN, 256, 0, stream>>>(Tbuf, Ybuf, nnb, body23, hfeat);
    // 7. residual block + collation + head
    {
        int total = BB * NN * CFEAT;
        k_res_linear<<<(total + 255) / 256, 256, 0, stream>>>(hfeat, res_W, res_b, lin);
    }
    k_bn_stats<<<1, 256, 0, stream>>>(lin, mv);
    {
        int total = BB * 2 * CFEAT;
        k_collate<<<(total + 255) / 256, 256, 0, stream>>>(hfeat, lin, mv, res_g, res_bt, xsum);
    }
    k_col_linear<<<(BB * 128 + 255) / 256, 256, 0, stream>>>(xsum, col_W, col_b, xc);
    k_bn2_stats<<<1, 128, 0, stream>>>(xc, mv2);
    k_final<<<1, BB, 0, stream>>>(xc, mv2, bn_g, bn_bt, out_W, out_b, out);
}
